// FactorisedTFSelfAttention_12524124635219
// MI455X (gfx1250) — matrix operand order repaired and checked by reading, not compiled, not run on hardware
//
#include <hip/hip_runtime.h>

typedef __attribute__((ext_vector_type(16))) _Float16 v16h;
typedef __attribute__((ext_vector_type(8)))  _Float16 v8h;
typedef __attribute__((ext_vector_type(8)))  float    v8f;

#define DEVINL __device__ __forceinline__

// ---------------------------------------------------------------- WMMA helpers
DEVINL v8f wmma32(v16h a, v16h b, v8f c) {
  // D = A(16x32 f16) * B(32x16 f16) + C(16x16 f32)
  return __builtin_amdgcn_wmma_f32_16x16x32_f16(false, a, false, b, (short)0, c,
                                                false, false);
}

DEVINL v16h cat8(v8h lo, v8h hi) {
  return __builtin_shufflevector(lo, hi, 0, 1, 2, 3, 4, 5, 6, 7, 8, 9, 10, 11,
                                 12, 13, 14, 15);
}

// A fragment (M=16 x K=32) from row-major f16 LDS [row][k].
// Lanes 0-15: row=lane, k 0..7 & 16..23; lanes 16-31: row=lane-16, k 8..15 &
// 24..31. stride in halves (multiple of 8), base 16B aligned.
DEVINL v16h ldsA(const _Float16* base, int stride, int lane) {
  int r  = lane & 15;
  int ko = (lane >> 4) << 3;  // 0 or 8
  const _Float16* p = base + r * stride + ko;
  return cat8(*(const v8h*)p, *(const v8h*)(p + 16));
}

// B fragment (K=32 x N=16) where B[k][n] = M[n][k], M row-major f16 LDS.
// Lanes 0-15: col n=lane, k 0..7 & 16..23; lanes 16-31: col n=lane-16, k 8..15 & 24..31.
DEVINL v16h ldsBt(const _Float16* base, int stride, int lane) {
  int n  = lane & 15;
  int ko = (lane >> 4) << 3;  // 0 or 8
  const _Float16* p = base + n * stride + ko;
  return cat8(*(const v8h*)p, *(const v8h*)(p + 16));
}

// convert 8 consecutive f32 -> f16 and store as one 16B LDS write
DEVINL void cvt_store8(_Float16* dst, const float* src) {
  float4 a = *(const float4*)src;
  float4 b = *(const float4*)(src + 4);
  v8h v;
  v[0] = (_Float16)a.x; v[1] = (_Float16)a.y;
  v[2] = (_Float16)a.z; v[3] = (_Float16)a.w;
  v[4] = (_Float16)b.x; v[5] = (_Float16)b.y;
  v[6] = (_Float16)b.z; v[7] = (_Float16)b.w;
  *(v8h*)dst = v;
}

DEVINL unsigned pack2(float lo, float hi) {
  union { _Float16 h[2]; unsigned u; } z;
  z.h[0] = (_Float16)lo;
  z.h[1] = (_Float16)hi;
  return z.u;
}

// ------------------------------------------------------------------ constants
#define BB   4
#define CC   512
#define HH   128
#define WW   256
#define HWSZ 32768  // H*W
#define INR  128

// =========================================================== K1: proj_in GEMM
// t_buf[b*HW + p][i] = sum_c x[b,c,p] * Win[i,c]   (M=131072, N=128, K=512)
__global__ __launch_bounds__(256) void proj_in_kernel(
    const float* __restrict__ x, const float* __restrict__ w,
    float* __restrict__ out) {
  __shared__ _Float16 xa[128 * 40];  // [pixel][c]  k-slice of A, f16
  __shared__ _Float16 wb[128 * 40];  // [o][c]      k-slice of W, f16
  const int tid = threadIdx.x, lane = tid & 31, wid = tid >> 5;
  const long gr0 = (long)blockIdx.x * 128;  // global pixel row
  const int b = (int)(gr0 >> 15);
  const int p0 = (int)(gr0 & (HWSZ - 1));
  const float* xb = x + (long)b * CC * HWSZ + p0;

  v8f acc[8] = {};
  for (int k0 = 0; k0 < CC; k0 += 32) {
    {  // stage x tile: 128 pixels x 32 channels (2 channels/thread, packed b32)
      int cp = (tid >> 4) * 2;
      int pp = (tid & 15) * 8;
      const float* s0 = xb + (long)(k0 + cp) * HWSZ + pp;
      const float* s1 = s0 + HWSZ;
      float4 a0 = *(const float4*)(s0), a1 = *(const float4*)(s0 + 4);
      float4 b0 = *(const float4*)(s1), b1 = *(const float4*)(s1 + 4);
      _Float16* dst = &xa[pp * 40 + cp];
      *(unsigned*)&dst[0 * 40] = pack2(a0.x, b0.x);
      *(unsigned*)&dst[1 * 40] = pack2(a0.y, b0.y);
      *(unsigned*)&dst[2 * 40] = pack2(a0.z, b0.z);
      *(unsigned*)&dst[3 * 40] = pack2(a0.w, b0.w);
      *(unsigned*)&dst[4 * 40] = pack2(a1.x, b1.x);
      *(unsigned*)&dst[5 * 40] = pack2(a1.y, b1.y);
      *(unsigned*)&dst[6 * 40] = pack2(a1.z, b1.z);
      *(unsigned*)&dst[7 * 40] = pack2(a1.w, b1.w);
    }
    {  // stage weights: 128 outputs x 32 channels (vectorized)
      int o = tid >> 1;
      int c0 = (tid & 1) * 16;
      const float* src = w + (long)o * CC + k0 + c0;
      cvt_store8(wb + o * 40 + c0, src);
      cvt_store8(wb + o * 40 + c0 + 8, src + 8);
    }
    __syncthreads();
    v16h a = ldsA(xa + wid * 16 * 40, 40, lane);
    v16h bfr[8];
#pragma unroll
    for (int nt = 0; nt < 8; ++nt) bfr[nt] = ldsBt(wb + nt * 16 * 40, 40, lane);
#pragma unroll
    for (int nt = 0; nt < 8; ++nt) acc[nt] = wmma32(a, bfr[nt], acc[nt]);
    __syncthreads();
  }
  int col = lane & 15, ro = (lane >> 4) * 8;
#pragma unroll
  for (int nt = 0; nt < 8; ++nt)
#pragma unroll
    for (int r = 0; r < 8; ++r)
      out[(gr0 + wid * 16 + ro + r) * INR + nt * 16 + col] = acc[nt][r];
}

// ================================================= K2/K3: fused MHSA (+LN, +res)
// One workgroup per (sequence, head). inBuf rows are contiguous per sequence.
// Output row for position m = b*HW + pos + m*outRowStride (transposed layout
// feeding the next stage).
template <int N>
__global__ __launch_bounds__(256) void attn_kernel(
    const float* __restrict__ inBuf, float* __restrict__ outBuf,
    const float* __restrict__ wq, const float* __restrict__ wk,
    const float* __restrict__ wv, const float* __restrict__ lnw,
    const float* __restrict__ lnb, int seqPerB, int outRowStride) {
  extern __shared__ char smem[];
  const int tid = threadIdx.x, lane = tid & 31, wid = tid >> 5;
  const int head = blockIdx.x & 3;
  const int seq = blockIdx.x >> 2;
  const int b = seq / seqPerB;
  const int pos = seq % seqPerB;
  const long outBase = (long)b * HWSZ + pos;

  constexpr int XSTR = 136;     // xn row stride (halves)
  constexpr int QSTR = 40;      // Q/K row stride
  constexpr int PSTR = N + 8;   // P / Vt row stride
  constexpr int WSTR = 136;     // staged weight row stride
  constexpr int NT = N / 16;    // S column tiles
  constexpr int KT = N / 32;    // PV contraction steps
  constexpr int RB = N / 128;   // row passes (8 waves x 16 rows)

  _Float16* xn = (_Float16*)smem;                  // N x XSTR (f16)
  _Float16* Ql = (_Float16*)(smem + (size_t)N * 272);
  _Float16* Kl = Ql + N * QSTR;
  _Float16* Vt = Kl + N * QSTR;                    // 32 x PSTR (transposed V)
  _Float16* Wl = Vt + 32 * PSTR;                   // 96 x WSTR (Wq|Wk|Wv head slice)
  _Float16* Pl = xn;                               // P overlays xn after QKV

  // ---- Phase 1a: stage this head's Q/K/V weight slices to LDS (f16) ----
  for (int rr = tid; rr < 96; rr += 256) {
    int t3 = rr >> 5;   // 0:q 1:k 2:v
    int i = rr & 31;
    const float* src =
        (t3 == 0 ? wq : (t3 == 1 ? wk : wv)) + (long)(head * 32 + i) * INR;
    _Float16* dst = Wl + rr * WSTR;
#pragma unroll
    for (int j = 0; j < INR; j += 8) cvt_store8(dst + j, src + j);
  }

  // ---- Phase 1b: LayerNorm each row -> xn (f16, vectorized stores) ----
  if (tid < N) {
    const float* row = inBuf + ((long)seq * N + tid) * INR;
    float s = 0.f, ss = 0.f;
#pragma unroll 8
    for (int j = 0; j < INR; j += 4) {
      float4 v = *(const float4*)(row + j);
      s += v.x + v.y + v.z + v.w;
      ss += v.x * v.x + v.y * v.y + v.z * v.z + v.w * v.w;
    }
    float mean = s * (1.f / INR);
    float var = ss * (1.f / INR) - mean * mean;
    float inv = rsqrtf(var + 1e-5f);
    _Float16* dst = xn + tid * XSTR;
#pragma unroll 4
    for (int j = 0; j < INR; j += 8) {
      float4 a = *(const float4*)(row + j);
      float4 c = *(const float4*)(row + j + 4);
      float4 wa = *(const float4*)(lnw + j);
      float4 wc = *(const float4*)(lnw + j + 4);
      float4 ba = *(const float4*)(lnb + j);
      float4 bc = *(const float4*)(lnb + j + 4);
      v8h v;
      v[0] = (_Float16)((a.x - mean) * inv * wa.x + ba.x);
      v[1] = (_Float16)((a.y - mean) * inv * wa.y + ba.y);
      v[2] = (_Float16)((a.z - mean) * inv * wa.z + ba.z);
      v[3] = (_Float16)((a.w - mean) * inv * wa.w + ba.w);
      v[4] = (_Float16)((c.x - mean) * inv * wc.x + bc.x);
      v[5] = (_Float16)((c.y - mean) * inv * wc.y + bc.y);
      v[6] = (_Float16)((c.z - mean) * inv * wc.z + bc.z);
      v[7] = (_Float16)((c.w - mean) * inv * wc.w + bc.w);
      *(v8h*)(dst + j) = v;
    }
  }
  __syncthreads();

  // ---- Phase 2: Q/K/V for this head (d = 32), weights from LDS ----
#pragma unroll
  for (int rb = 0; rb < RB; ++rb) {
    int mrow = rb * 128 + wid * 16;
    const _Float16* xrow = xn + mrow * XSTR;
#pragma unroll
    for (int ct = 0; ct < 2; ++ct) {
      v8f aQ = {}, aK = {}, aV = {};
#pragma unroll
      for (int kt = 0; kt < 4; ++kt) {
        v16h a  = ldsA(xrow + kt * 32, XSTR, lane);
        v16h bq = ldsBt(Wl + (ct * 16) * WSTR + kt * 32, WSTR, lane);
        v16h bk = ldsBt(Wl + (32 + ct * 16) * WSTR + kt * 32, WSTR, lane);
        v16h bv = ldsBt(Wl + (64 + ct * 16) * WSTR + kt * 32, WSTR, lane);
        aQ = wmma32(a, bq, aQ);
        aK = wmma32(a, bk, aK);
        aV = wmma32(a, bv, aV);
      }
      int col = lane & 15, ro = (lane >> 4) * 8;
#pragma unroll
      for (int r = 0; r < 8; ++r) {
        int m = mrow + ro + r;
        Ql[m * QSTR + ct * 16 + col] = (_Float16)aQ[r];
        Kl[m * QSTR + ct * 16 + col] = (_Float16)aK[r];
      }
      v8h vv;  // V transposed: 8 consecutive seq positions -> one 16B store
#pragma unroll
      for (int r = 0; r < 8; ++r) vv[r] = (_Float16)aV[r];
      *(v8h*)&Vt[(ct * 16 + col) * PSTR + mrow + ro] = vv;
    }
  }
  __syncthreads();

  // ---- Phase 3: S = QK^T, softmax, O = P V, residual, transposed store ----
  const float scale = 0.17677669529663687f;  // 1/sqrt(32)
  _Float16* Pw = Pl + wid * 16 * PSTR;       // per-wave P region
  for (int rb = 0; rb < RB; ++rb) {
    int mrow = rb * 128 + wid * 16;
    v16h aQ = ldsA(Ql + mrow * QSTR, QSTR, lane);
    v8f s[NT];
#pragma unroll
    for (int g = 0; g < NT; g += 4) {  // batch B-frag loads in clauses of 4
      v16h bf[4];
#pragma unroll
      for (int q = 0; q < 4; ++q)
        bf[q] = ldsBt(Kl + (g + q) * 16 * QSTR, QSTR, lane);
#pragma unroll
      for (int q = 0; q < 4; ++q) {
        v8f z = {};
        s[g + q] = wmma32(aQ, bf[q], z);
      }
    }
    float lrow[8];
#pragma unroll
    for (int r = 0; r < 8; ++r) {
      float mx = -3.0e38f;
#pragma unroll
      for (int nt = 0; nt < NT; ++nt) mx = fmaxf(mx, s[nt][r] * scale);
#pragma unroll
      for (int msk = 1; msk < 16; msk <<= 1)
        mx = fmaxf(mx, __shfl_xor(mx, msk, 32));  // row lives in a 16-lane half
      float l = 0.f;
#pragma unroll
      for (int nt = 0; nt < NT; ++nt) {
        float e = __expf(s[nt][r] * scale - mx);
        s[nt][r] = e;
        l += e;
      }
#pragma unroll
      for (int msk = 1; msk < 16; msk <<= 1) l += __shfl_xor(l, msk, 32);
      lrow[r] = l;
    }
    // stage P (f16) so it can re-enter WMMA in A-layout
    int col = lane & 15, ro = (lane >> 4) * 8;
#pragma unroll
    for (int nt = 0; nt < NT; ++nt)
#pragma unroll
      for (int r = 0; r < 8; ++r)
        Pw[(ro + r) * PSTR + nt * 16 + col] = (_Float16)s[nt][r];
    __builtin_amdgcn_wave_barrier();
    asm volatile("" ::: "memory");

    v8f o0 = {}, o1 = {};
#pragma unroll
    for (int kt = 0; kt < KT; kt += 2) {
      v16h aP0 = ldsA(Pw + kt * 32, PSTR, lane);
      v16h aP1 = ldsA(Pw + (kt + 1) * 32, PSTR, lane);
      v16h b00 = ldsBt(Vt + kt * 32, PSTR, lane);
      v16h b01 = ldsBt(Vt + 16 * PSTR + kt * 32, PSTR, lane);
      v16h b10 = ldsBt(Vt + (kt + 1) * 32, PSTR, lane);
      v16h b11 = ldsBt(Vt + 16 * PSTR + (kt + 1) * 32, PSTR, lane);
      o0 = wmma32(aP0, b00, o0);
      o1 = wmma32(aP0, b01, o1);
      o0 = wmma32(aP1, b10, o0);
      o1 = wmma32(aP1, b11, o1);
    }
#pragma unroll
    for (int r = 0; r < 8; ++r) {
      int m = mrow + ro + r;
      float il = 1.f / lrow[r];
      long orow = (outBase + (long)m * outRowStride) * INR;
      int c0 = head * 32 + col;
      const float* res = inBuf + ((long)seq * N + m) * INR;
      outBuf[orow + c0] = res[c0] + o0[r] * il;
      outBuf[orow + c0 + 16] = res[c0 + 16] + o1[r] * il;
    }
  }
}

// ====================================== K4: proj_out GEMM + GN partial sums
// y[b,o,p] (channel-major) = sum_i g[b*HW+p][i] * Wout[o,i]
__global__ __launch_bounds__(256) void proj_out_kernel(
    const float* __restrict__ g, const float* __restrict__ w,
    float* __restrict__ y, float* __restrict__ psum, float* __restrict__ psq) {
  extern __shared__ char sm4[];
  _Float16* xa = (_Float16*)sm4;       // 128 x 40
  _Float16* wb = xa + 128 * 40;        // 128 x 40
  float* tr = (float*)(wb + 128 * 40); // transpose buffer: [o][p] 128 x 132
  float* redS = tr + 128 * 132;        // 64
  float* redQ = redS + 64;             // 64

  const int tid = threadIdx.x, lane = tid & 31, wid = tid >> 5;
  const int pixTile = blockIdx.x;  // 0..1023
  const int colTile = blockIdx.y;  // 0..3
  const long gr0 = (long)pixTile * 128;
  const int b = pixTile >> 8;      // 256 pixel tiles per batch
  const int pt = pixTile & 255;
  const int p0 = (int)(gr0 & (HWSZ - 1));
  const int o0 = colTile * 128;

  v8f acc[8] = {};
  for (int k0 = 0; k0 < INR; k0 += 32) {
    {
      int r = tid >> 1, c0 = (tid & 1) * 16;
      const float* src = g + (gr0 + r) * INR + k0 + c0;
      cvt_store8(xa + r * 40 + c0, src);
      cvt_store8(xa + r * 40 + c0 + 8, src + 8);
      const float* ws = w + (long)(o0 + r) * INR + k0 + c0;
      cvt_store8(wb + r * 40 + c0, ws);
      cvt_store8(wb + r * 40 + c0 + 8, ws + 8);
    }
    __syncthreads();
    v16h a = ldsA(xa + wid * 16 * 40, 40, lane);
    v16h bfr[8];
#pragma unroll
    for (int nt = 0; nt < 8; ++nt) bfr[nt] = ldsBt(wb + nt * 16 * 40, 40, lane);
#pragma unroll
    for (int nt = 0; nt < 8; ++nt) acc[nt] = wmma32(a, bfr[nt], acc[nt]);
    __syncthreads();
  }

  int col = lane & 15, ro = (lane >> 4) * 8;
#pragma unroll
  for (int nt = 0; nt < 8; ++nt) {  // each 16-col tile == one GN group slice
    float sv = 0.f, sq = 0.f;
#pragma unroll
    for (int r = 0; r < 8; ++r) {
      float v = acc[nt][r];
      sv += v;
      sq += v * v;
    }
    float* td = &tr[(nt * 16 + col) * 132 + wid * 16 + ro];
    *(float4*)(td) = make_float4(acc[nt][0], acc[nt][1], acc[nt][2], acc[nt][3]);
    *(float4*)(td + 4) =
        make_float4(acc[nt][4], acc[nt][5], acc[nt][6], acc[nt][7]);
#pragma unroll
    for (int msk = 1; msk < 32; msk <<= 1) {
      sv += __shfl_xor(sv, msk, 32);
      sq += __shfl_xor(sq, msk, 32);
    }
    if (lane == 0) {
      redS[wid * 8 + nt] = sv;
      redQ[wid * 8 + nt] = sq;
    }
  }
  __syncthreads();
  if (tid < 8) {  // deterministic per-(b,group) partials
    float sv = 0.f, sq = 0.f;
    for (int wv = 0; wv < 8; ++wv) {
      sv += redS[wv * 8 + tid];
      sq += redQ[wv * 8 + tid];
    }
    long gi = ((long)b * 32 + colTile * 8 + tid) * 256 + pt;
    psum[gi] = sv;
    psq[gi] = sq;
  }
  // coalesced channel-major writeout through the LDS transpose
  int o = tid >> 1, ph = (tid & 1) * 64;
  float* dst = y + (long)b * CC * HWSZ + (long)(o0 + o) * HWSZ + p0 + ph;
  const float* srcT = tr + o * 132 + ph;
#pragma unroll
  for (int j = 0; j < 64; j += 4) *(float4*)(dst + j) = *(const float4*)(srcT + j);
}

// ============================================================ K5: GN stats
__global__ __launch_bounds__(256) void gn_stats_kernel(
    const float* __restrict__ psum, const float* __restrict__ psq,
    float* __restrict__ stats) {
  __shared__ float sS[256], sQ[256];
  int bg = blockIdx.x, t = threadIdx.x;
  sS[t] = psum[(long)bg * 256 + t];
  sQ[t] = psq[(long)bg * 256 + t];
  __syncthreads();
  for (int off = 128; off; off >>= 1) {
    if (t < off) {
      sS[t] += sS[t + off];
      sQ[t] += sQ[t + off];
    }
    __syncthreads();
  }
  if (t == 0) {
    const float n = 16.f * (float)HWSZ;
    float m = sS[0] / n;
    float v = sQ[0] / n - m * m;
    stats[bg * 2] = m;
    stats[bg * 2 + 1] = rsqrtf(v + 1e-5f);
  }
}

// ================================================ K6: GN apply + residual
__global__ __launch_bounds__(256) void finalize_kernel(
    const float* __restrict__ x, const float* __restrict__ y,
    const float* __restrict__ stats, const float* __restrict__ gnw,
    const float* __restrict__ gnb, float* __restrict__ out) {
  long i4 = (long)blockIdx.x * blockDim.x + threadIdx.x;  // float4 index
  int c = (int)((i4 >> 13) & (CC - 1));  // HW/4 = 8192 per channel
  int b = (int)(i4 >> 22);               // C*HW/4 per batch
  int bg = b * 32 + (c >> 4);
  float m = stats[bg * 2], inv = stats[bg * 2 + 1];
  float wsc = inv * gnw[c];
  float bb = gnb[c] - m * wsc;
  float4 xv = ((const float4*)x)[i4];
  float4 yv = ((const float4*)y)[i4];
  float4 o;
  o.x = xv.x + yv.x * wsc + bb;
  o.y = xv.y + yv.y * wsc + bb;
  o.z = xv.z + yv.z * wsc + bb;
  o.w = xv.w + yv.w * wsc + bb;
  ((float4*)out)[i4] = o;
}

// ================================================================= launcher
extern "C" void kernel_launch(void* const* d_in, const int* in_sizes, int n_in,
                              void* d_out, int out_size, void* d_ws,
                              size_t ws_size, hipStream_t stream) {
  (void)in_sizes; (void)n_in; (void)out_size; (void)ws_size;
  const float* x   = (const float*)d_in[0];
  const float* win = (const float*)d_in[1];
  const float* tq  = (const float*)d_in[2];
  const float* tk  = (const float*)d_in[3];
  const float* tv  = (const float*)d_in[4];
  const float* tnw = (const float*)d_in[5];
  const float* tnb = (const float*)d_in[6];
  const float* fq  = (const float*)d_in[7];
  const float* fk  = (const float*)d_in[8];
  const float* fv  = (const float*)d_in[9];
  const float* fnw = (const float*)d_in[10];
  const float* fnb = (const float*)d_in[11];
  const float* wout= (const float*)d_in[12];
  const float* gnw = (const float*)d_in[13];
  const float* gnb = (const float*)d_in[14];
  float* out = (float*)d_out;

  char* ws = (char*)d_ws;
  float* t_buf = (float*)(ws + 0);            // 64 MiB: (B*HW) x 128
  float* f_buf = (float*)(ws + 67108864);     // 64 MiB
  float* y_buf = (float*)(ws + 134217728);    // 256 MiB channel-major
  float* psum  = (float*)(ws + 402653184);    // 128 KiB
  float* psq   = (float*)(ws + 402784256);    // 128 KiB
  float* stats = (float*)(ws + 402915328);    // 1 KiB

  // K1: proj_in (WMMA f16, f32 accum)
  proj_in_kernel<<<1024, 256, 0, stream>>>(x, win, t_buf);

  // K2: temporal attention, 512 seqs of N=256, 4 heads -> f layout
  size_t sm256 = (size_t)256 * 272 + 2 * (size_t)256 * 80 + 64 * (256 + 8) +
                 96 * 136 * 2;
  attn_kernel<256><<<512 * 4, 256, sm256, stream>>>(t_buf, f_buf, tq, tk, tv,
                                                    tnw, tnb, HH, HH);

  // K3: frequency attention, 1024 seqs of N=128 -> pixel-major (aliases t_buf)
  size_t sm128 = (size_t)128 * 272 + 2 * (size_t)128 * 80 + 64 * (128 + 8) +
                 96 * 136 * 2;
  attn_kernel<128><<<1024 * 4, 256, sm128, stream>>>(f_buf, t_buf, fq, fk, fv,
                                                     fnw, fnb, WW, WW);

  // K4: proj_out + deterministic GN partials
  proj_out_kernel<<<dim3(1024, 4), 256, 88576, stream>>>(t_buf, wout, y_buf,
                                                         psum, psq);
  // K5: GN stats
  gn_stats_kernel<<<128, 256, 0, stream>>>(psum, psq, stats);
  // K6: GN apply + residual
  finalize_kernel<<<65536, 256, 0, stream>>>(x, y_buf, stats, gnw, gnb, out);
}
